// EdgeAttentionLayer_70789650972913
// MI455X (gfx1250) — matrix operand order repaired and checked by reading, not compiled, not run on hardware
//
#include <hip/hip_runtime.h>
#include <hip/hip_bf16.h>

// ---------------- problem constants ----------------
#define NN 50000
#define EE 800000
#define DD 128
#define FF 16
#define HH 8
#define HDIM 16
#define TILES_M (NN / 16)   // 3125

typedef __attribute__((ext_vector_type(16))) __bf16 v16bf;
typedef __attribute__((ext_vector_type(8)))  float  v8f;

// TDM descriptor group types (amdgpu-toolchain 6-arg builtin form)
typedef __attribute__((ext_vector_type(4))) unsigned int u32x4;
typedef __attribute__((ext_vector_type(8))) int          i32x8;
typedef __attribute__((ext_vector_type(4))) int          i32x4;

// ---------------- conversion / init kernels ----------------

__global__ void convert_x(const float* __restrict__ x,
                          __bf16* __restrict__ xb) {
    int i = blockIdx.x * blockDim.x + threadIdx.x;
    int n = NN * DD;
    for (; i < n; i += gridDim.x * blockDim.x) {
        xb[i] = (__bf16)x[i];
    }
}

__global__ void convert_weights(const float* __restrict__ Wq,
                                const float* __restrict__ Wk,
                                const float* __restrict__ Wv,
                                const float* __restrict__ Wo,
                                __bf16* __restrict__ Wb,    // [3][D][D]
                                __bf16* __restrict__ Wob) { // [D][D]
    int i = blockIdx.x * blockDim.x + threadIdx.x;
    if (i < DD * DD) {
        Wb[i]               = (__bf16)Wq[i];
        Wb[DD * DD + i]     = (__bf16)Wk[i];
        Wb[2 * DD * DD + i] = (__bf16)Wv[i];
        Wob[i]              = (__bf16)Wo[i];
    }
}

__global__ void init_node_accum(unsigned* __restrict__ nmax,
                                float* __restrict__ ssum,
                                float* __restrict__ asum) {
    int i = blockIdx.x * blockDim.x + threadIdx.x;
    if (i < NN * HH) {
        // ordered-uint encoding of -inf: bits=0xFF800000 (negative) -> ~bits
        nmax[i] = 0x007FFFFFu;
        ssum[i] = 0.0f;
        asum[i] = 0.0f;
    }
}

// ---------------- WMMA GEMM: Q/K/V = x @ W^T ----------------
// grid (TILES_M, 3), block 256 (8 waves). Wave w computes output tile
// rows [16*bx, 16*bx+16), cols [16*w, 16*w+16). K-loop: 4 x wmma 16x16x32.
// The 16x128 bf16 x-tile is 16 consecutive rows of a row-major matrix ->
// one contiguous 4KB chunk, staged into LDS by the Tensor Data Mover.

__global__ void qkv_gemm(const __bf16* __restrict__ xb,
                         const __bf16* __restrict__ Wb,
                         float* __restrict__ Q,
                         float* __restrict__ K,
                         float* __restrict__ V) {
    __shared__ __bf16 lds_a[16 * DD];

    const int mat   = blockIdx.y;
    const __bf16* W = Wb + mat * DD * DD;
    float* outp = (mat == 0) ? Q : ((mat == 1) ? K : V);

    const int row0 = blockIdx.x * 16;

    // --- TDM: DMA the contiguous 16x128 bf16 tile (4KB) into LDS ---
    if (threadIdx.x < 32) {   // wave 0 issues the descriptor
        unsigned long long ga = (unsigned long long)(uintptr_t)(xb + (size_t)row0 * DD);
        unsigned lds_off = (unsigned)(uintptr_t)&lds_a[0];

        u32x4 g0;
        g0[0] = 1u;                                  // count=1, is_restore=0
        g0[1] = lds_off;                             // lds_addr
        g0[2] = (unsigned)(ga & 0xFFFFFFFFu);        // global_addr[31:0]
        g0[3] = (unsigned)((ga >> 32) & 0x01FFFFFFu) // global_addr[56:32]
              | (2u << 30);                          // type=2 ("image")

        const unsigned tile0  = 16 * DD;             // 2048 bf16 elements
        const unsigned tens0  = (unsigned)(NN * DD); // tensor extent (elements)
        i32x8 g1;
        g1[0] = (int)(1u << 16);                     // data_size=1 -> 2 bytes
        g1[1] = (int)((tens0 & 0xFFFFu) << 16);      // tensor_dim0[15:0]
        g1[2] = (int)(((tens0 >> 16) & 0xFFFFu)      // tensor_dim0[31:16]
              | (1u << 16));                         // tensor_dim1 = 1
        g1[3] = (int)(tile0 << 16);                  // tile_dim0 = 2048
        g1[4] = 1;                                   // tile_dim1 = 1
        g1[5] = (int)tens0;                          // tensor_dim0_stride lo32
        g1[6] = 0;
        g1[7] = 0;

        i32x4 z4 = {0, 0, 0, 0};                     // groups 2/3 unused (<=2D)
        i32x8 z8 = {0, 0, 0, 0, 0, 0, 0, 0};
        __builtin_amdgcn_tensor_load_to_lds(g0, g1, z4, z4, z8, 0);
        __builtin_amdgcn_s_wait_tensorcnt(0);
    }
    __syncthreads();

    const int wave = threadIdx.x >> 5;        // 0..7 -> output col tile
    const int lane = threadIdx.x & 31;
    const int col0 = wave * 16;

    const int mrow  = lane & 15;              // A-frag: M = lane%16
    const int khalf = (lane >> 4) * 8;        // unused; A and B loads both use kb16
    const int ncol  = lane & 15;              // B/C/D: N = lane%16
    const int kb16  = (lane >> 4) * 16;       // B-frag K offset per lane half

    v8f acc = {};
    for (int k0 = 0; k0 < DD; k0 += 32) {
        v16bf a, b;
#pragma unroll
        for (int j = 0; j < 16; ++j) {
            a[j] = lds_a[mrow * DD + k0 + kb16 + j];
        }
#pragma unroll
        for (int j = 0; j < 16; ++j) {
            // B[k][n] = W[n][k]  (x @ W^T)
            b[j] = W[(col0 + ncol) * DD + (k0 + kb16 + j)];
        }
        acc = __builtin_amdgcn_wmma_f32_16x16x32_bf16(
                  false, a, false, b, (short)0, acc, false, false);
    }

#pragma unroll
    for (int r = 0; r < 8; ++r) {
        int m = r + 8 * (lane >> 4);
        outp[(row0 + m) * DD + col0 + ncol] = acc[r];
    }
}

// ---------------- per-edge attention scores + segment max ----------------
// one thread per (edge, head); 8 consecutive threads cover one edge's
// 512B Q/K rows -> coalesced, L2-resident gathers.

__global__ void attn_score(const float* __restrict__ Q,
                           const float* __restrict__ K,
                           const int* __restrict__ src,
                           const int* __restrict__ tgt,
                           const float* __restrict__ eattr,
                           const float* __restrict__ We,
                           float* __restrict__ attn,
                           unsigned* __restrict__ nmax) {
    int t = blockIdx.x * blockDim.x + threadIdx.x;
    if (t >= EE * HH) return;
    int e = t >> 3, h = t & 7;
    int s = src[e], g = tgt[e];

    const float* q  = Q + (size_t)s * DD + h * HDIM;
    const float* kk = K + (size_t)g * DD + h * HDIM;
    float dot = 0.0f;
#pragma unroll
    for (int i = 0; i < HDIM; ++i) dot += q[i] * kk[i];

    const float* ea = eattr + (size_t)e * FF;
    float bias = 0.0f;
#pragma unroll
    for (int f = 0; f < FF; ++f) bias += ea[f] * We[h * FF + f];

    float a = dot * 0.25f + bias;   // 1/sqrt(16) = 0.25
    attn[t] = a;

    unsigned bits = __float_as_uint(a);
    unsigned key  = (bits & 0x80000000u) ? ~bits : (bits | 0x80000000u);
    atomicMax(&nmax[(size_t)g * HH + h], key);
}

// ---------------- exp(attn - max) + segment sum ----------------

__global__ void attn_expsum(const int* __restrict__ tgt,
                            float* __restrict__ attn,
                            const unsigned* __restrict__ nmax,
                            float* __restrict__ ssum) {
    int t = blockIdx.x * blockDim.x + threadIdx.x;
    if (t >= EE * HH) return;
    int e = t >> 3, h = t & 7;
    int g = tgt[e];

    unsigned key  = nmax[(size_t)g * HH + h];
    unsigned bits = (key & 0x80000000u) ? (key & 0x7FFFFFFFu) : ~key;
    float m = __uint_as_float(bits);
    if (m < -3.0e38f) m = 0.0f;     // mirrors reference isneginf guard

    float ex = __expf(attn[t] - m);
    attn[t] = ex;
    atomicAdd(&ssum[(size_t)g * HH + h], ex);
}

// ---------------- normalized-weight sum per (node, head) ----------------
// Since the reference scatters a[e,h] * V[tgt[e]] into tgt[e], V factors
// out of each node's sum: agg[n,h,:] = V[n,h,:] * sum_{tgt=n} a[e,h].
// Only the scalar weight sum needs atomics (16x fewer than vector scatter).

__global__ void asum_accum(const int* __restrict__ tgt,
                           const float* __restrict__ attn,
                           const float* __restrict__ ssum,
                           float* __restrict__ asum) {
    int t = blockIdx.x * blockDim.x + threadIdx.x;
    if (t >= EE * HH) return;
    int e = t >> 3, h = t & 7;
    int g = tgt[e];

    float s = ssum[(size_t)g * HH + h];
    if (s == 0.0f) s = 1.0f;
    atomicAdd(&asum[(size_t)g * HH + h], attn[t] / s);
}

// ---------------- WMMA GEMM: out = (asum ⊙ V) @ Wo^T + bo ----------------
// The asum * V product is fused into the LDS staging (f32 -> bf16).

__global__ void out_gemm(const float* __restrict__ V,
                         const float* __restrict__ asum,
                         const __bf16* __restrict__ Wob,
                         const float* __restrict__ bo,
                         float* __restrict__ out) {
    __shared__ __bf16 lds_a[16 * DD];

    const int row0 = blockIdx.x * 16;
    for (int i = threadIdx.x; i < 16 * DD; i += 256) {
        int r = i >> 7, c = i & (DD - 1);
        int node = row0 + r;
        float w = asum[(size_t)node * HH + (c >> 4)];
        lds_a[i] = (__bf16)(V[(size_t)node * DD + c] * w);
    }
    __syncthreads();

    const int wave = threadIdx.x >> 5;
    const int lane = threadIdx.x & 31;
    const int col0 = wave * 16;

    const int mrow  = lane & 15;
    const int khalf = (lane >> 4) * 8;
    const int ncol  = lane & 15;
    const int kb16  = (lane >> 4) * 16;

    v8f acc = {};
    for (int k0 = 0; k0 < DD; k0 += 32) {
        v16bf a, b;
#pragma unroll
        for (int j = 0; j < 16; ++j) {
            a[j] = lds_a[mrow * DD + k0 + kb16 + j];
        }
#pragma unroll
        for (int j = 0; j < 16; ++j) {
            b[j] = Wob[(col0 + ncol) * DD + (k0 + kb16 + j)];
        }
        acc = __builtin_amdgcn_wmma_f32_16x16x32_bf16(
                  false, a, false, b, (short)0, acc, false, false);
    }

    float bias = bo[col0 + ncol];
#pragma unroll
    for (int r = 0; r < 8; ++r) {
        int m = r + 8 * (lane >> 4);
        out[(row0 + m) * DD + col0 + ncol] = acc[r] + bias;
    }
}

// ---------------- launch ----------------

extern "C" void kernel_launch(void* const* d_in, const int* in_sizes, int n_in,
                              void* d_out, int out_size, void* d_ws, size_t ws_size,
                              hipStream_t stream) {
    const float* x     = (const float*)d_in[0];
    const int*   eidx  = (const int*)d_in[1];
    const float* eattr = (const float*)d_in[2];
    const float* Wq    = (const float*)d_in[3];
    const float* Wk    = (const float*)d_in[4];
    const float* Wv    = (const float*)d_in[5];
    const float* We    = (const float*)d_in[6];
    const float* Wo    = (const float*)d_in[7];
    const float* bo    = (const float*)d_in[8];
    float* out = (float*)d_out;

    const int* src = eidx;
    const int* tgt = eidx + EE;

    // workspace partition (all re-initialized each call -> deterministic)
    char* p = (char*)d_ws;
    auto take = [&](size_t bytes) -> char* {
        char* r = p;
        p += (bytes + 255) & ~(size_t)255;
        return r;
    };
    float*    Qb   = (float*)   take((size_t)NN * DD * 4);
    float*    Kb   = (float*)   take((size_t)NN * DD * 4);
    float*    Vb   = (float*)   take((size_t)NN * DD * 4);
    float*    attn = (float*)   take((size_t)EE * HH * 4);
    unsigned* nmax = (unsigned*)take((size_t)NN * HH * 4);
    float*    ssum = (float*)   take((size_t)NN * HH * 4);
    float*    asum = (float*)   take((size_t)NN * HH * 4);
    __bf16*   xb   = (__bf16*)  take((size_t)NN * DD * 2);
    __bf16*   Wb   = (__bf16*)  take((size_t)3 * DD * DD * 2);
    __bf16*   Wob  = (__bf16*)  take((size_t)DD * DD * 2);
    (void)ws_size; (void)in_sizes; (void)n_in; (void)out_size;

    const int nd_blocks = (NN * DD + 255) / 256;        // 25000
    const int eh_blocks = (EE * HH + 255) / 256;        // 25000
    const int nh_blocks = (NN * HH + 255) / 256;

    convert_x<<<nd_blocks, 256, 0, stream>>>(x, xb);
    convert_weights<<<(DD * DD + 255) / 256, 256, 0, stream>>>(Wq, Wk, Wv, Wo, Wb, Wob);
    init_node_accum<<<nh_blocks, 256, 0, stream>>>(nmax, ssum, asum);

    qkv_gemm<<<dim3(TILES_M, 3), 256, 0, stream>>>(xb, Wb, Qb, Kb, Vb);

    attn_score<<<eh_blocks, 256, 0, stream>>>(Qb, Kb, src, tgt, eattr, We, attn, nmax);
    attn_expsum<<<eh_blocks, 256, 0, stream>>>(tgt, attn, nmax, ssum);
    asum_accum<<<eh_blocks, 256, 0, stream>>>(tgt, attn, ssum, asum);

    out_gemm<<<TILES_M, 256, 0, stream>>>(Vb, asum, Wob, bo, out);
}
